// Model_83837761618370
// MI455X (gfx1250) — matrix operand order repaired and checked by reading, not compiled, not run on hardware
//
#include <hip/hip_runtime.h>
#include <stdint.h>

// ---------------- problem constants ----------------
constexpr int Bn   = 16;
constexpr int Sx   = 720;
constexpr int Cc   = 862;
constexpr int Pp   = 24;
constexpr int LQ   = 30;
constexpr int DMn  = 128;
constexpr int PRED = 336;
constexpr int BC   = Bn * Cc;          // 13792 (divisible by 16)
constexpr int SPAD = 736;              // 720 padded to 23*32
constexpr int CPAD = 864;              // 862 padded to 54*16
constexpr int NBLK = (BC + 127) / 128; // 108 column blocks for PEC kernel
constexpr int MT   = PRED / 16;        // 21
constexpr int NT   = CPAD / 16;        // 54

typedef __bf16 v16bf __attribute__((ext_vector_type(16)));
typedef float  v8f   __attribute__((ext_vector_type(8)));

union Frag { uint32_t u[8]; uint4 q[2]; v16bf v; };
union Acc  { float    f[8]; v8f   v; };

__device__ __forceinline__ uint16_t f2bf(float f) {
    uint32_t x = __float_as_uint(f);
    uint32_t r = (x + 0x7FFFu + ((x >> 16) & 1u)) >> 16;  // round-to-nearest-even
    return (uint16_t)r;
}

// branch-free GELU (tanh/sigmoid form): x * sigmoid(1.5957691*x*(1+0.044715 x^2))
// lowers to v_exp_f32 + v_rcp_f32, no divergence; plenty accurate vs bf16 WMMA.
__device__ __forceinline__ float gelu_fast(float x) {
    float z = 1.595769122f * x * fmaf(0.044715f * x, x, 1.0f);
    float e = __builtin_amdgcn_exp2f(-1.442695041f * z);
    return x * __builtin_amdgcn_rcpf(1.0f + e);
}

// ---------------- kernel 1: RevIN stats + normalize + build Xp -------------
// one wave per (b,c); Xp layout [p][bc][32] bf16, K (=q) contiguous, q=30,31 zero
__global__ __launch_bounds__(256) void revin_kernel(
    const float* __restrict__ x, const float* __restrict__ rev_w,
    const float* __restrict__ rev_b, float* __restrict__ meanv,
    float* __restrict__ stdv, uint16_t* __restrict__ Xp)
{
    int w = threadIdx.x >> 5, lane = threadIdx.x & 31;
    int bc = blockIdx.x * 8 + w;
    if (bc >= BC) return;
    int b = bc / Cc, c = bc % Cc;
    const float* xp = x + (size_t)b * Sx * Cc + c;

    float s = 0.f, s2 = 0.f;
    for (int t = lane; t < Sx; t += 32) {
        float v = xp[(size_t)t * Cc];
        s += v; s2 += v * v;
    }
    #pragma unroll
    for (int off = 16; off; off >>= 1) {
        s  += __shfl_xor(s,  off, 32);
        s2 += __shfl_xor(s2, off, 32);
    }
    float mean = s * (1.0f / (float)Sx);
    float var  = s2 * (1.0f / (float)Sx) - mean * mean;
    float sd   = sqrtf(var + 1e-5f);
    if (lane == 0) { meanv[bc] = mean; stdv[bc] = sd; }

    float rw = rev_w[c], rb = rev_b[c], isd = 1.0f / sd;
    for (int t = lane; t < Sx; t += 32) {
        float v  = xp[(size_t)t * Cc];
        float xn = (v - mean) * isd * rw + rb;
        int q = t / Pp, pp = t % Pp;
        Xp[((size_t)pp * BC + bc) * 32 + q] = f2bf(xn);
    }
    if (lane < Pp) {  // zero K-pad rows q=30,31
        Xp[((size_t)lane * BC + bc) * 32 + 30] = 0;
        Xp[((size_t)lane * BC + bc) * 32 + 31] = 0;
    }
}

// ---------------- kernel 2: pack layer-(EL-1) weights + proj_w to bf16 -----
__global__ __launch_bounds__(256) void prep_weights(
    const float* __restrict__ W1, const float* __restrict__ W2,
    const float* __restrict__ pw, uint16_t* __restrict__ W1b,
    uint16_t* __restrict__ W2b, uint16_t* __restrict__ projb)
{
    int tid = blockIdx.x * 256 + threadIdx.x;
    if (tid < Pp * DMn * 32) {                    // W1b[p][d][32]
        int p = tid / (DMn * 32), r = tid % (DMn * 32);
        int d = r / 32, q = r % 32;
        float v = (q < LQ) ? W1[(((size_t)Pp + p) * DMn + d) * LQ + q] : 0.f; // layer 1
        W1b[tid] = f2bf(v);
    }
    if (tid < Pp * 32 * DMn) {                    // W2b[p][q(32)][d]
        int p = tid / (32 * DMn), r = tid % (32 * DMn);
        int q = r / DMn, d = r % DMn;
        float v = (q < LQ) ? W2[(((size_t)Pp + p) * LQ + q) * DMn + d] : 0.f; // layer 1
        W2b[tid] = f2bf(v);
    }
    if (tid < PRED * SPAD) {                      // projb[m][736]
        int m = tid / SPAD, k = tid % SPAD;
        float v = (k < Sx) ? pw[(size_t)m * Sx + k] : 0.f;
        projb[tid] = f2bf(v);
    }
}

__global__ __launch_bounds__(256) void zero_u32(uint32_t* __restrict__ p, int n) {
    int i = blockIdx.x * 256 + threadIdx.x;
    if (i < n) p[i] = 0u;
}

// ---------------- kernel 3: fused PEC block (layer EL-1 only) --------------
// Y_p = W2_p * gelu(W1_p * X_p); one block = one p, 128 bc-columns; 8 waves.
__global__ __launch_bounds__(256) void pec_kernel(
    const uint16_t* __restrict__ Xp,   // [24][BC][32]
    const uint16_t* __restrict__ W1b,  // [24][128][32]
    const uint16_t* __restrict__ W2b,  // [24][32][128]
    uint16_t* __restrict__ yT)         // [16][864][736]
{
    __shared__ uint16_t Hl[128 * 128]; // H staged col-major: Hl[col][row] bf16
    int p    = blockIdx.x / NBLK;
    int nb   = blockIdx.x % NBLK;
    int lane = threadIdx.x & 31;
    int w    = threadIdx.x >> 5;
    int l15  = lane & 15;
    int hi   = lane >> 4;

    int col = nb * 128 + w * 16 + l15;           // this lane's B-fragment column

    // B fragment (X_p): K = hi*16 + 2v, pairs contiguous in Xp
    Frag bx;
    if (col < BC) {
        const uint4* src =
            (const uint4*)(Xp + ((size_t)p * BC + col) * 32 + hi * 16);
        bx.q[0] = src[0];
        bx.q[1] = src[1];
    } else {
        #pragma unroll
        for (int v = 0; v < 8; ++v) bx.u[v] = 0u;
    }

    // GEMM1: 8 M-tiles of W1_p (128 rows), K=32 -> one WMMA each
    int cl = w * 16 + l15;                       // local LDS column
    #pragma unroll
    for (int mt = 0; mt < 8; ++mt) {
        Frag a;
        const uint16_t* wsrc = W1b + ((size_t)p * DMn + mt * 16 + l15) * 32;
        #pragma unroll
        for (int v = 0; v < 8; ++v) {
            int k = hi * 16 + 2 * v;
            a.u[v] = *(const uint32_t*)(wsrc + k);
        }
        Acc acc;
        #pragma unroll
        for (int j = 0; j < 8; ++j) acc.f[j] = 0.f;
        acc.v = __builtin_amdgcn_wmma_f32_16x16x32_bf16(
            false, a.v, false, bx.v, (short)0, acc.v, false, false);

        // GELU + packed store: lane's 8 rows (mt*16 + hi*8 + 0..7) are
        // contiguous in Hl[col][*] -> one 16B LDS store.
        uint32_t pk[4];
        #pragma unroll
        for (int j = 0; j < 4; ++j) {
            uint32_t lo = f2bf(gelu_fast(acc.f[2 * j]));
            uint32_t hh = f2bf(gelu_fast(acc.f[2 * j + 1]));
            pk[j] = lo | (hh << 16);
        }
        *(uint4*)(Hl + (size_t)cl * 128 + mt * 16 + hi * 8) =
            make_uint4(pk[0], pk[1], pk[2], pk[3]);
    }
    __syncthreads();

    // GEMM2: M=32 (2 tiles of W2_p rows q), K=128 (4 steps), wave w -> N tile w
    Acc acc0, acc1;
    #pragma unroll
    for (int j = 0; j < 8; ++j) { acc0.f[j] = 0.f; acc1.f[j] = 0.f; }
    #pragma unroll
    for (int kk = 0; kk < 4; ++kk) {
        int k0 = kk * 32;
        Frag bh;                                  // B frag from LDS H (32B contig)
        const uint4* hsrc = (const uint4*)(Hl + (size_t)cl * 128 + k0 + hi * 16);
        bh.q[0] = hsrc[0];
        bh.q[1] = hsrc[1];
        #pragma unroll
        for (int mt = 0; mt < 2; ++mt) {
            Frag a;
            const uint16_t* wsrc = W2b + ((size_t)p * 32 + mt * 16 + l15) * DMn + k0;
            #pragma unroll
            for (int v = 0; v < 8; ++v) {
                int k = hi * 16 + 2 * v;
                a.u[v] = *(const uint32_t*)(wsrc + k);
            }
            if (mt == 0)
                acc0.v = __builtin_amdgcn_wmma_f32_16x16x32_bf16(
                    false, a.v, false, bh.v, (short)0, acc0.v, false, false);
            else
                acc1.v = __builtin_amdgcn_wmma_f32_16x16x32_bf16(
                    false, a.v, false, bh.v, (short)0, acc1.v, false, false);
        }
    }

    // scatter-store: yT[b][c][s = q*24 + p], rows q<30 valid
    if (col < BC) {
        int b = col / Cc, c = col % Cc;
        uint16_t* dst = yT + ((size_t)b * CPAD + c) * SPAD;
        #pragma unroll
        for (int j = 0; j < 8; ++j) {
            int q0 = j + hi * 8;                 // 0..15, always < 30
            dst[q0 * Pp + p] = f2bf(acc0.f[j]);
            int q1 = 16 + j + hi * 8;
            if (q1 < LQ) dst[q1 * Pp + p] = f2bf(acc1.f[j]);
        }
    }
}

// ---------------- kernel 4: projection GEMM + RevIN denorm -----------------
// one wave per 16x16 output tile: out[b][m][c] = sum_s projb[m][s]*yT[b][c][s]
__global__ __launch_bounds__(256) void proj_kernel(
    const uint16_t* __restrict__ projb, // [336][736]
    const uint16_t* __restrict__ yT,    // [16][864][736]
    const float* __restrict__ proj_b, const float* __restrict__ rev_w,
    const float* __restrict__ rev_b, const float* __restrict__ meanv,
    const float* __restrict__ stdv, float* __restrict__ out)
{
    int w = threadIdx.x >> 5, lane = threadIdx.x & 31;
    int t = blockIdx.x * 8 + w;
    if (t >= Bn * MT * NT) return;
    int b  = t / (MT * NT);
    int r  = t % (MT * NT);
    int mt = r / NT, nt = r % NT;
    int l15 = lane & 15, hi = lane >> 4;

    Acc acc;
    #pragma unroll
    for (int j = 0; j < 8; ++j) acc.f[j] = 0.f;

    const uint16_t* arow = projb + ((size_t)(mt * 16 + l15)) * SPAD;
    const uint16_t* brow = yT + ((size_t)b * CPAD + nt * 16 + l15) * SPAD;
    for (int kk = 0; kk < SPAD / 32; ++kk) {     // 23 K-steps
        int k0 = kk * 32;
        Frag a, bb;
        #pragma unroll
        for (int v = 0; v < 8; ++v) {
            int ka = k0 + hi * 16 + 2 * v;
            a.u[v] = *(const uint32_t*)(arow + ka);
        }
        const uint4* bsrc = (const uint4*)(brow + k0 + hi * 16);
        bb.q[0] = bsrc[0];
        bb.q[1] = bsrc[1];
        acc.v = __builtin_amdgcn_wmma_f32_16x16x32_bf16(
            false, a.v, false, bb.v, (short)0, acc.v, false, false);
    }

    int c = nt * 16 + l15;
    if (c < Cc) {
        float rw = rev_w[c], rb = rev_b[c];
        float mn = meanv[b * Cc + c], sd = stdv[b * Cc + c];
        float inv = 1.0f / (rw + 1e-10f);        // EPS*EPS
        #pragma unroll
        for (int j = 0; j < 8; ++j) {
            int m = mt * 16 + j + hi * 8;        // always < 336
            float v = acc.f[j] + proj_b[m];
            v = (v - rb) * inv;
            v = v * sd + mn;
            out[((size_t)b * PRED + m) * Cc + c] = v;
        }
    }
}

// ---------------- host launcher --------------------------------------------
extern "C" void kernel_launch(void* const* d_in, const int* in_sizes, int n_in,
                              void* d_out, int out_size, void* d_ws, size_t ws_size,
                              hipStream_t stream) {
    const float* x   = (const float*)d_in[0];
    const float* W1  = (const float*)d_in[1];
    const float* W2  = (const float*)d_in[2];
    const float* pw  = (const float*)d_in[3];
    const float* pb  = (const float*)d_in[4];
    const float* rw  = (const float*)d_in[5];
    const float* rb  = (const float*)d_in[6];
    float* out = (float*)d_out;

    char* ws = (char*)d_ws;
    size_t off = 0;
    auto alloc = [&](size_t bytes) -> void* {
        void* p = ws + off;
        off = (off + bytes + 255) & ~(size_t)255;
        return p;
    };
    float*    meanv = (float*)alloc((size_t)BC * 4);
    float*    stdv  = (float*)alloc((size_t)BC * 4);
    uint16_t* Xp    = (uint16_t*)alloc((size_t)Pp * BC * 32 * 2);
    uint16_t* W1b   = (uint16_t*)alloc((size_t)Pp * DMn * 32 * 2);
    uint16_t* W2b   = (uint16_t*)alloc((size_t)Pp * 32 * DMn * 2);
    uint16_t* projb = (uint16_t*)alloc((size_t)PRED * SPAD * 2);
    uint16_t* yT    = (uint16_t*)alloc((size_t)Bn * CPAD * SPAD * 2);

    // 1) RevIN + Xp build (BC/8 = 1724 exact)
    revin_kernel<<<BC / 8, 256, 0, stream>>>(x, rw, rb, meanv, stdv, Xp);

    // 2) weight packing (covers max range 336*736 = 247296 = 966*256)
    prep_weights<<<(PRED * SPAD + 255) / 256, 256, 0, stream>>>(
        W1, W2, pw, W1b, W2b, projb);

    // 3) zero yT (so K/N pad regions read as 0 in projection)
    {
        int n = (Bn * CPAD * SPAD) / 2;          // uint32 count
        zero_u32<<<(n + 255) / 256, 256, 0, stream>>>((uint32_t*)yT, n);
    }

    // 4) fused PEC layer (24 periods x 108 column blocks)
    pec_kernel<<<Pp * NBLK, 256, 0, stream>>>(Xp, W1b, W2b, yT);

    // 5) projection + denorm (16*21*54 = 18144 tiles / 8 waves = 2268 blocks)
    proj_kernel<<<(Bn * MT * NT) / 8, 256, 0, stream>>>(
        projb, yT, pb, rw, rb, meanv, stdv, out);
}
